// BiLSTM_CRF_49675591746101
// MI455X (gfx1250) — compile-verified, matrix operand order checked by reading, not run on hardware
//
#include <hip/hip_runtime.h>

// ---------------- problem constants ----------------
#define Bsz 64
#define Tlen 256
#define Lch 16
#define Ktag 12
#define CEdim 64
#define CHH 64           // char hidden per dir
#define CGATE 256        // 4*CHH
#define WINdim 640       // WE + CH
#define WHH 256          // word hidden per dir
#define WGATE 1024       // 4*WHH
#define NSEQ (Bsz*Tlen)  // 16384
#define MT_W (NSEQ/16)   // 1024 word M-tiles
#define NEGV -10000.0f

// ---------------- bf16 helpers ----------------
typedef __bf16 bf16_t;
typedef __attribute__((ext_vector_type(16))) __bf16 v16bf;
typedef __attribute__((ext_vector_type(8)))  float  v8f;

__device__ __forceinline__ unsigned short f2bf(float f) {
  unsigned u = __float_as_uint(f);
  unsigned r = u + 0x7FFFu + ((u >> 16) & 1u);
  return (unsigned short)(r >> 16);
}
__device__ __forceinline__ float bf2f(unsigned short h) {
  return __uint_as_float(((unsigned)h) << 16);
}
// fast transcendentals: v_exp_f32 + v_rcp_f32, no IEEE-div expansion
__device__ __forceinline__ float sigm(float x) {
  return __builtin_amdgcn_rcpf(1.0f + __expf(-x));
}
__device__ __forceinline__ float tanh_fast(float x) {
  return 1.0f - 2.0f * __builtin_amdgcn_rcpf(1.0f + __expf(2.0f * x));
}

union TileU { uint4 q[2]; v16bf v; };
union US8u  { uint4 q; unsigned short u[8]; };

// Load a 16x32 bf16 tile (A layout; also used for B since W is stored [N][K]):
// lane l: row = l&15 ; K-halves: lanes<16 -> K {0..7,16..23}, lanes>=16 -> K {8..15,24..31}
__device__ __forceinline__ v16bf load_tile(const unsigned short* base, int stride) {
  const int lane = threadIdx.x & 31;
  const unsigned short* p = base + (lane & 15) * stride + ((lane >> 4) << 3);
  TileU t;
  t.q[0] = *(const uint4*)(p);
  t.q[1] = *(const uint4*)(p + 16);
  return t.v;
}

__device__ __forceinline__ v8f wmma_bf16(v16bf a, v16bf b, v8f c) {
  return __builtin_amdgcn_wmma_f32_16x16x32_bf16(false, a, false, b, (short)0, c, false, false);
}

// ---------------- small utility kernels ----------------
__global__ void k_f2bf(const float* __restrict__ src, unsigned short* __restrict__ dst, int n) {
  int i = blockIdx.x * 256 + threadIdx.x;
  if (i < n) dst[i] = f2bf(src[i]);
}

// Wtag [12][512] -> padded bf16 [16][512], rows 12..15 zero
__global__ void k_wtag_pad(const float* __restrict__ w, unsigned short* __restrict__ dst) {
  int n = blockIdx.x;               // 0..15
  int k = threadIdx.x;              // 0..511
  dst[n * 512 + k] = (n < Ktag) ? f2bf(w[n * 512 + k]) : (unsigned short)0;
}

// x[:,128:640] = bf16(word_embed[sentence])
__global__ void k_gather_word(const int* __restrict__ sentence, const float* __restrict__ we,
                              unsigned short* __restrict__ x) {
  int seq = blockIdx.x;
  int idx = sentence[seq];
  const float* src = we + (size_t)idx * 512;
  unsigned short* dst = x + (size_t)seq * WINdim + 128;
  for (int j = threadIdx.x; j < 512; j += 256) dst[j] = f2bf(src[j]);
}

// ---------------- char BiLSTM ----------------
// 1024 blocks x 512 threads (16 waves). Each block: 16 sequences.
// fwd: 16 steps, keep last h -> x[:,0:64]; bwd: 1 step on last char -> x[:,64:128]
__global__ void k_char_lstm(const int* __restrict__ chars, const unsigned short* __restrict__ cemb,
                            const unsigned short* __restrict__ wih_f, const unsigned short* __restrict__ whh_f,
                            const float* __restrict__ b_f,
                            const unsigned short* __restrict__ wih_b, const unsigned short* __restrict__ whh_b,
                            const float* __restrict__ b_b,
                            const float* __restrict__ h0c, const float* __restrict__ c0c,
                            unsigned short* __restrict__ x) {
  __shared__ __align__(16) unsigned short s_x[16 * CEdim];
  __shared__ __align__(16) unsigned short s_h[16 * CHH];
  __shared__ float s_c[16 * CHH];
  __shared__ float s_g[16 * CGATE];

  const int s0 = blockIdx.x * 16;
  const int tid = threadIdx.x;
  const int w = tid >> 5;
  const int col = tid & 15;
  const int rhi = ((tid >> 4) & 1) << 3;

  // fwd init
  for (int i = tid; i < 16 * CHH; i += 512) {
    int seq = s0 + (i >> 6), j = i & 63;
    s_h[i] = f2bf(h0c[(size_t)seq * CHH + j]);
    s_c[i] = c0c[(size_t)seq * CHH + j];
  }
  __syncthreads();

  for (int t = 0; t < Lch; ++t) {
    for (int i = tid; i < 16 * CEdim; i += 512) {
      int row = i >> 6, j = i & 63;
      int ci = chars[(s0 + row) * Lch + t];
      s_x[i] = cemb[ci * CEdim + j];
    }
    __syncthreads();
    // gate tile: wave w owns cols [w*16, w*16+16)
    v8f c;
    float bias = b_f[w * 16 + col];
#pragma unroll
    for (int v = 0; v < 8; ++v) c[v] = bias;
#pragma unroll
    for (int k = 0; k < 2; ++k)
      c = wmma_bf16(load_tile(s_x + k * 32, CEdim),
                    load_tile(wih_f + w * 16 * CEdim + k * 32, CEdim), c);
#pragma unroll
    for (int k = 0; k < 2; ++k)
      c = wmma_bf16(load_tile(s_h + k * 32, CHH),
                    load_tile(whh_f + w * 16 * CHH + k * 32, CHH), c);
#pragma unroll
    for (int v = 0; v < 8; ++v) s_g[(v + rhi) * CGATE + w * 16 + col] = c[v];
    __syncthreads();
    for (int i = tid; i < 16 * CHH; i += 512) {
      int row = i >> 6, j = i & 63;
      float gi = s_g[row * CGATE + j], gf = s_g[row * CGATE + 64 + j];
      float gg = s_g[row * CGATE + 128 + j], go = s_g[row * CGATE + 192 + j];
      float cn = sigm(gf) * s_c[i] + sigm(gi) * tanh_fast(gg);
      s_c[i] = cn;
      s_h[i] = f2bf(sigm(go) * tanh_fast(cn));
    }
    __syncthreads();
  }
  // write fwd output (last h) -> x[:,0:64]
  for (int i = tid; i < 16 * CHH; i += 512) {
    int row = i >> 6, j = i & 63;
    x[(size_t)(s0 + row) * WINdim + j] = s_h[i];
  }
  __syncthreads();

  // bwd: one step on char at position L-1
  for (int i = tid; i < 16 * CHH; i += 512) {
    int seq = s0 + (i >> 6), j = i & 63;
    s_h[i] = f2bf(h0c[(size_t)NSEQ * CHH + (size_t)seq * CHH + j]);
    s_c[i] = c0c[(size_t)NSEQ * CHH + (size_t)seq * CHH + j];
  }
  __syncthreads();
  for (int i = tid; i < 16 * CEdim; i += 512) {
    int row = i >> 6, j = i & 63;
    int ci = chars[(s0 + row) * Lch + (Lch - 1)];
    s_x[i] = cemb[ci * CEdim + j];
  }
  __syncthreads();
  {
    v8f c;
    float bias = b_b[w * 16 + col];
#pragma unroll
    for (int v = 0; v < 8; ++v) c[v] = bias;
#pragma unroll
    for (int k = 0; k < 2; ++k)
      c = wmma_bf16(load_tile(s_x + k * 32, CEdim),
                    load_tile(wih_b + w * 16 * CEdim + k * 32, CEdim), c);
#pragma unroll
    for (int k = 0; k < 2; ++k)
      c = wmma_bf16(load_tile(s_h + k * 32, CHH),
                    load_tile(whh_b + w * 16 * CHH + k * 32, CHH), c);
#pragma unroll
    for (int v = 0; v < 8; ++v) s_g[(v + rhi) * CGATE + w * 16 + col] = c[v];
  }
  __syncthreads();
  for (int i = tid; i < 16 * CHH; i += 512) {
    int row = i >> 6, j = i & 63;
    float gi = s_g[row * CGATE + j], gf = s_g[row * CGATE + 64 + j];
    float gg = s_g[row * CGATE + 128 + j], go = s_g[row * CGATE + 192 + j];
    float cn = sigm(gf) * s_c[i] + sigm(gi) * tanh_fast(gg);
    x[(size_t)(s0 + row) * WINdim + 64 + j] = f2bf(sigm(go) * tanh_fast(cn));
  }
}

// ---------------- word input projection ----------------
// G stored in WMMA-accumulator tile layout:
//   tile tau = ((dir*MT_W + mtile)*64 + ntile); element (tau*32 + lane)*8 + v  (bf16)
// so the recurrent kernel initializes each accumulator with ONE b128 load per lane.
// grid (1024 Mtiles, 8 Ngroups, 2 dirs), block 256 (8 waves, 1 N-tile each)
__global__ void k_word_inproj(const unsigned short* __restrict__ x,
                              const unsigned short* __restrict__ wih_f,
                              const unsigned short* __restrict__ wih_b,
                              unsigned short* __restrict__ G) {
  __shared__ __align__(16) unsigned short s_a[16 * WINdim]; // 20KB
  const int m = blockIdx.x, ng = blockIdx.y, dir = blockIdx.z;
  const unsigned short* wih = dir ? wih_b : wih_f;
  const int r0 = m * 16, t = r0 >> 6, b0 = r0 & 63;
  for (int i = threadIdx.x; i < 16 * WINdim; i += 256) {
    int row = i / WINdim, cc = i % WINdim;
    s_a[i] = x[((size_t)(b0 + row) * Tlen + t) * WINdim + cc];
  }
  __syncthreads();
  const int w = threadIdx.x >> 5;
  const int lane = threadIdx.x & 31;
  const int ntile = ng * 8 + w;
  const int n0 = ntile * 16;
  v8f c;
#pragma unroll
  for (int v = 0; v < 8; ++v) c[v] = 0.0f;
#pragma unroll
  for (int k = 0; k < WINdim / 32; ++k)
    c = wmma_bf16(load_tile(s_a + k * 32, WINdim),
                  load_tile(wih + (size_t)n0 * WINdim + k * 32, WINdim), c);
  US8u s;
#pragma unroll
  for (int v = 0; v < 8; ++v) s.u[v] = f2bf(c[v]);
  const size_t tau = ((size_t)dir * MT_W + m) * 64 + ntile;
  *(uint4*)(G + (tau * 32 + lane) * 8) = s.q;
}

// ---------------- word recurrent LSTM (one WG per direction) ----------------
// block 1024 threads = 32 waves; each wave: 2 tasks; task = (batch-tile m, hidden-tile jt)
// owning the i/f/g/o gate tiles -> elementwise update entirely in accumulator layout.
__global__ void k_word_rnn(const unsigned short* __restrict__ G,
                           const unsigned short* __restrict__ whh_f,
                           const unsigned short* __restrict__ whh_b,
                           const float* __restrict__ wb_f, const float* __restrict__ wb_b,
                           const float* __restrict__ h0w, const float* __restrict__ c0w,
                           unsigned short* __restrict__ hcat) {
  extern __shared__ unsigned char smem[];
  unsigned short* s_h = (unsigned short*)smem;            // [64][256] bf16
  float* s_c = (float*)(smem + Bsz * WHH * 2);            // [64][256] f32
  const int dir = blockIdx.x;
  const unsigned short* whh = dir ? whh_b : whh_f;
  const float* wb = dir ? wb_b : wb_f;
  const int tid = threadIdx.x;
  for (int i = tid; i < Bsz * WHH; i += 1024) {
    s_h[i] = f2bf(h0w[(size_t)dir * Bsz * WHH + i]);
    s_c[i] = c0w[(size_t)dir * Bsz * WHH + i];
  }
  __syncthreads();
  const int w = tid >> 5;
  const int lane = tid & 31;
  const int col = tid & 15;
  const int rhi = ((tid >> 4) & 1) << 3;
  v8f C[2][4];
  for (int s = 0; s < Tlen; ++s) {
    const int t = dir ? (Tlen - 1 - s) : s;
    // phase 1: GEMM (reads s_h)
    for (int q = 0; q < 2; ++q) {
      const int task = w * 2 + q;
      const int m = task >> 4, jt = task & 15;
      for (int g = 0; g < 4; ++g) {
        const int ntile = g * 16 + jt;
        const int n0 = ntile * 16;
        // accumulator init: one b128 load of the pre-packed G tile + bias
        const size_t tau = ((size_t)dir * MT_W + (t * 4 + m)) * 64 + ntile;
        US8u gi;
        gi.q = *(const uint4*)(G + (tau * 32 + lane) * 8);
        v8f c;
        const float bias = wb[n0 + col];
#pragma unroll
        for (int v = 0; v < 8; ++v) c[v] = bf2f(gi.u[v]) + bias;
#pragma unroll
        for (int k = 0; k < WHH / 32; ++k)
          c = wmma_bf16(load_tile(s_h + m * 16 * WHH + k * 32, WHH),
                        load_tile(whh + (size_t)ntile * 16 * WHH + k * 32, WHH), c);
        C[q][g] = c;
      }
    }
    __syncthreads();
    // phase 2: elementwise LSTM update in accumulator layout (writes s_h/s_c only)
    for (int q = 0; q < 2; ++q) {
      const int task = w * 2 + q;
      const int m = task >> 4, jt = task & 15;
#pragma unroll
      for (int v = 0; v < 8; ++v) {
        const int brow = m * 16 + v + rhi;
        const int hid = jt * 16 + col;
        const float ci = C[q][0][v], cf = C[q][1][v], cg = C[q][2][v], co = C[q][3][v];
        const float cn = sigm(cf) * s_c[brow * WHH + hid] + sigm(ci) * tanh_fast(cg);
        s_c[brow * WHH + hid] = cn;
        s_h[brow * WHH + hid] = f2bf(sigm(co) * tanh_fast(cn));
      }
    }
    __syncthreads();
    // phase 3a: prefetch next step's 128KB G block into WGP cache while VMEM is idle
    if (s + 1 < Tlen) {
      const int tn = dir ? (t - 1) : (t + 1);
      const unsigned short* gnext =
          G + (((size_t)dir * MT_W + tn * 4) * 64) * 256;  // 4 mtiles * 64 ntiles * 256 el
      __builtin_prefetch(gnext + (size_t)tid * 128, 0, 3); // 1024 threads x 256B stride
    }
    // phase 3b: vectorized copy s_h -> hcat[:, t] (b128 LDS loads + b128 global stores);
    // overlaps next step's phase 1 (both only READ s_h); phase-1-end barrier fences it.
    for (int i = tid * 8; i < Bsz * WHH; i += 1024 * 8) {
      const int brow = i >> 8, hid = i & 255;
      uint4 hv = *(const uint4*)(s_h + i);
      *(uint4*)(hcat + ((size_t)(brow * Tlen + t)) * 512 + dir * WHH + hid) = hv;
    }
  }
}

// ---------------- feats = hcat @ Wtag^T + btag ----------------
__global__ void k_feats(const unsigned short* __restrict__ hcat,
                        const unsigned short* __restrict__ wtagp,
                        const float* __restrict__ btag, float* __restrict__ feats) {
  const int w = threadIdx.x >> 5;
  const int mt = blockIdx.x * 8 + w;
  const int col = threadIdx.x & 15;
  const int rhi = ((threadIdx.x >> 4) & 1) << 3;
  v8f c;
  const float bias = (col < Ktag) ? btag[col] : 0.0f;
#pragma unroll
  for (int v = 0; v < 8; ++v) c[v] = bias;
#pragma unroll
  for (int k = 0; k < 512 / 32; ++k)
    c = wmma_bf16(load_tile(hcat + (size_t)mt * 16 * 512 + k * 32, 512),
                  load_tile(wtagp + k * 32, 512), c);
#pragma unroll
  for (int v = 0; v < 8; ++v)
    feats[(size_t)(mt * 16 + v + rhi) * 16 + col] = c[v];
}

// ---------------- CRF ----------------
__global__ void k_crf_gold(const float* __restrict__ feats, const int* __restrict__ tags,
                           const float* __restrict__ trans, float* __restrict__ gold) {
  const int b = blockIdx.x, lane = threadIdx.x;
  float acc = 0.0f;
  for (int t = lane; t < Tlen; t += 32) {
    const int cur = tags[b * Tlen + t];
    const int prev = (t == 0) ? 0 : tags[b * Tlen + t - 1];
    acc += feats[((size_t)b * Tlen + t) * 16 + cur] + trans[cur * Ktag + prev];
  }
#pragma unroll
  for (int off = 16; off > 0; off >>= 1) acc += __shfl_xor(acc, off, 32);
  if (lane == 0) gold[b] = acc + trans[2 * Ktag + tags[b * Tlen + Tlen - 1]];
}

__global__ void k_crf_fwd(const float* __restrict__ feats, const float* __restrict__ trans,
                          float* __restrict__ fwdv) {
  const int b = blockIdx.x, lane = threadIdx.x;
  __shared__ float s_alpha[16];
  if (lane < 16) s_alpha[lane] = (lane == 0) ? 0.0f : NEGV;
  float trj[Ktag];
  if (lane < Ktag) {
#pragma unroll
    for (int i = 0; i < Ktag; ++i) trj[i] = trans[lane * Ktag + i];
  }
  __syncthreads();
  for (int t = 0; t < Tlen; ++t) {
    float na = NEGV;
    if (lane < Ktag) {
      float m = -1e30f;
#pragma unroll
      for (int i = 0; i < Ktag; ++i) m = fmaxf(m, s_alpha[i] + trj[i]);
      float ss = 0.0f;
#pragma unroll
      for (int i = 0; i < Ktag; ++i) ss += __expf(s_alpha[i] + trj[i] - m);
      na = m + __logf(ss) + feats[((size_t)b * Tlen + t) * 16 + lane];
    }
    __syncthreads();
    if (lane < Ktag) s_alpha[lane] = na;
    __syncthreads();
  }
  float v = (lane < Ktag) ? (s_alpha[lane] + trans[2 * Ktag + lane]) : -1e30f;
  float m = v;
#pragma unroll
  for (int off = 16; off > 0; off >>= 1) m = fmaxf(m, __shfl_xor(m, off, 32));
  float e = (lane < Ktag) ? __expf(v - m) : 0.0f;
#pragma unroll
  for (int off = 16; off > 0; off >>= 1) e += __shfl_xor(e, off, 32);
  if (lane == 0) fwdv[b] = m + __logf(e);
}

__global__ void k_final(const float* __restrict__ fwdv, const float* __restrict__ gold,
                        float* __restrict__ out) {
  __shared__ float red[64];
  const int t = threadIdx.x;
  red[t] = fwdv[t] - gold[t];
  __syncthreads();
  for (int s = 32; s > 0; s >>= 1) {
    if (t < s) red[t] += red[t + s];
    __syncthreads();
  }
  if (t == 0) out[0] = red[0] / (float)Bsz;
}

// ---------------- host launcher ----------------
extern "C" void kernel_launch(void* const* d_in, const int* in_sizes, int n_in,
                              void* d_out, int out_size, void* d_ws, size_t ws_size,
                              hipStream_t stream) {
  (void)in_sizes; (void)n_in; (void)out_size; (void)ws_size;
  const int*   sentence  = (const int*)d_in[0];
  const int*   chars     = (const int*)d_in[1];
  const int*   tags      = (const int*)d_in[2];
  const float* word_emb  = (const float*)d_in[3];
  const float* char_emb  = (const float*)d_in[4];
  const float* cWih_f = (const float*)d_in[5];
  const float* cWhh_f = (const float*)d_in[6];
  const float* cb_f   = (const float*)d_in[7];
  const float* cWih_b = (const float*)d_in[8];
  const float* cWhh_b = (const float*)d_in[9];
  const float* cb_b   = (const float*)d_in[10];
  const float* wWih_f = (const float*)d_in[11];
  const float* wWhh_f = (const float*)d_in[12];
  const float* wb_f   = (const float*)d_in[13];
  const float* wWih_b = (const float*)d_in[14];
  const float* wWhh_b = (const float*)d_in[15];
  const float* wb_b   = (const float*)d_in[16];
  const float* Wtag   = (const float*)d_in[17];
  const float* btag   = (const float*)d_in[18];
  const float* trans  = (const float*)d_in[19];
  const float* h0c    = (const float*)d_in[20];
  const float* c0c    = (const float*)d_in[21];
  const float* h0w    = (const float*)d_in[22];
  const float* c0w    = (const float*)d_in[23];
  float* out = (float*)d_out;

  char* base = (char*)d_ws;
  size_t off = 0;
  auto take = [&](size_t bytes) -> char* {
    char* r = base + off;
    off += (bytes + 255) & ~(size_t)255;
    return r;
  };
  unsigned short* ws_cwih_f = (unsigned short*)take(CGATE * CEdim * 2);
  unsigned short* ws_cwhh_f = (unsigned short*)take(CGATE * CHH * 2);
  unsigned short* ws_cwih_b = (unsigned short*)take(CGATE * CEdim * 2);
  unsigned short* ws_cwhh_b = (unsigned short*)take(CGATE * CHH * 2);
  unsigned short* ws_cemb   = (unsigned short*)take(128 * CEdim * 2);
  unsigned short* ws_wwih_f = (unsigned short*)take((size_t)WGATE * WINdim * 2);
  unsigned short* ws_wwhh_f = (unsigned short*)take((size_t)WGATE * WHH * 2);
  unsigned short* ws_wwih_b = (unsigned short*)take((size_t)WGATE * WINdim * 2);
  unsigned short* ws_wwhh_b = (unsigned short*)take((size_t)WGATE * WHH * 2);
  unsigned short* ws_wtagp  = (unsigned short*)take(16 * 512 * 2);
  unsigned short* ws_x      = (unsigned short*)take((size_t)NSEQ * WINdim * 2);
  unsigned short* ws_G      = (unsigned short*)take((size_t)2 * NSEQ * WGATE * 2);
  unsigned short* ws_hcat   = (unsigned short*)take((size_t)NSEQ * 512 * 2);
  float* ws_feats = (float*)take((size_t)NSEQ * 16 * 4);
  float* ws_gold  = (float*)take(Bsz * 4);
  float* ws_fwd   = (float*)take(Bsz * 4);

  auto cv = [&](const float* s, unsigned short* d, int n) {
    k_f2bf<<<(n + 255) / 256, 256, 0, stream>>>(s, d, n);
  };
  cv(cWih_f, ws_cwih_f, CGATE * CEdim);
  cv(cWhh_f, ws_cwhh_f, CGATE * CHH);
  cv(cWih_b, ws_cwih_b, CGATE * CEdim);
  cv(cWhh_b, ws_cwhh_b, CGATE * CHH);
  cv(char_emb, ws_cemb, 128 * CEdim);
  cv(wWih_f, ws_wwih_f, WGATE * WINdim);
  cv(wWhh_f, ws_wwhh_f, WGATE * WHH);
  cv(wWih_b, ws_wwih_b, WGATE * WINdim);
  cv(wWhh_b, ws_wwhh_b, WGATE * WHH);
  k_wtag_pad<<<16, 512, 0, stream>>>(Wtag, ws_wtagp);

  k_gather_word<<<NSEQ, 256, 0, stream>>>(sentence, word_emb, ws_x);
  k_char_lstm<<<NSEQ / 16, 512, 0, stream>>>(chars, ws_cemb,
      ws_cwih_f, ws_cwhh_f, cb_f, ws_cwih_b, ws_cwhh_b, cb_b, h0c, c0c, ws_x);
  k_word_inproj<<<dim3(MT_W, 8, 2), 256, 0, stream>>>(ws_x, ws_wwih_f, ws_wwih_b, ws_G);
  k_word_rnn<<<2, 1024, Bsz * WHH * 2 + Bsz * WHH * 4, stream>>>(
      ws_G, ws_wwhh_f, ws_wwhh_b, wb_f, wb_b, h0w, c0w, ws_hcat);
  k_feats<<<NSEQ / 16 / 8, 256, 0, stream>>>(ws_hcat, ws_wtagp, btag, ws_feats);
  k_crf_gold<<<Bsz, 32, 0, stream>>>(ws_feats, tags, trans, ws_gold);
  k_crf_fwd<<<Bsz, 32, 0, stream>>>(ws_feats, trans, ws_fwd);
  k_final<<<1, 64, 0, stream>>>(ws_fwd, ws_gold, out);
}
